// VectorQuantizer_24060406792913
// MI455X (gfx1250) — matrix operand order repaired and checked by reading, not compiled, not run on hardware
//
#include <hip/hip_runtime.h>

#define DIM    256
#define KCODES 8192
#define MTOT   32768          // BATCH * SEQ
#define TM     128            // query rows per block (8 waves x 16)
#define TN     32             // codebook rows per LDS tile
#define NTILES (KCODES / TN)  // 256
#define TILE_BYTES (TN * DIM * 2)   // 16384 B of bf16 per tile

typedef __attribute__((ext_vector_type(16))) __bf16 v16bf;
typedef __attribute__((ext_vector_type(8)))  __bf16 v8bf;
typedef __attribute__((ext_vector_type(8)))  float  v8f;

// ---------------------------------------------------------------------------
// Kernel 1: codebook f32 -> bf16 (workspace) + ||c||^2 per code row.
// One wave (32 lanes) per codebook row; lane covers 8 consecutive floats.
// ---------------------------------------------------------------------------
__global__ __launch_bounds__(256) void vq_prep(const float* __restrict__ cb,
                                               unsigned short* __restrict__ cb_bf_u,
                                               float* __restrict__ csq) {
  __bf16* cb_bf = (__bf16*)cb_bf_u;
  const int lane = threadIdx.x & 31;
  const int wave = threadIdx.x >> 5;
  const int row  = blockIdx.x * 8 + wave;

  const float* src = cb + (size_t)row * DIM + lane * 8;
  float4 f0 = *(const float4*)(src);
  float4 f1 = *(const float4*)(src + 4);

  float p = f0.x * f0.x + f0.y * f0.y + f0.z * f0.z + f0.w * f0.w
          + f1.x * f1.x + f1.y * f1.y + f1.z * f1.z + f1.w * f1.w;
#pragma unroll
  for (int s = 16; s > 0; s >>= 1) p += __shfl_xor(p, s, 32);
  if (lane == 0) csq[row] = p;

  v8bf vb;
  vb[0] = (__bf16)f0.x; vb[1] = (__bf16)f0.y; vb[2] = (__bf16)f0.z; vb[3] = (__bf16)f0.w;
  vb[4] = (__bf16)f1.x; vb[5] = (__bf16)f1.y; vb[6] = (__bf16)f1.z; vb[7] = (__bf16)f1.w;
  *(v8bf*)(cb_bf + (size_t)row * DIM + lane * 8) = vb;
}

// ---------------------------------------------------------------------------
// Kernel 2: main VQ search.
// LDS B-tile layout is K-major: Bs[buf][cc][row][16 bf16], cc = ks*2 + half,
// so each WMMA B fragment read is contiguous 32B per lane, conflict-free.
// Tiles are filled with global_load_async_to_lds_b128 (ASYNCcnt, no VGPR data)
// doing the global(row-major) -> LDS(K-major) transpose for free.
// ---------------------------------------------------------------------------
__global__ __launch_bounds__(256) void vq_main(const float* __restrict__ x,
                                               const float* __restrict__ cb,
                                               const unsigned short* __restrict__ cb_bf_u,
                                               const float* __restrict__ csq,
                                               float* __restrict__ out_idx,
                                               float* __restrict__ out_q) {
  __shared__ __align__(32) __bf16 Bs[2][16][TN][16];  // 32 KB (2 x 16 KB tiles)
  __shared__ float csq_s[KCODES];                     // 32 KB

  const int tid   = threadIdx.x;
  const int wave  = tid >> 5;
  const int lane  = tid & 31;
  const int half  = lane >> 4;     // 0: low 16 lanes, 1: high 16 lanes
  const int rl    = lane & 15;
  const int rbase = blockIdx.x * TM + wave * 16;

  // ---- stage csq into LDS (once) ----
  {
    const float4* cg = (const float4*)csq;
    float4*       cl = (float4*)csq_s;
#pragma unroll
    for (int i = 0; i < 8; ++i) cl[i * 256 + tid] = cg[i * 256 + tid];
  }

  // ---- async-copy addressing: thread owns 4 rows x one 16B unit ----
  // global unit:  tile*16K + row*512 + u*16   (row-major bf16 codebook)
  // LDS unit:     buf*16K + ((u>>1)*TN + row)*32 + (u&1)*16   (K-major)
  const unsigned u    = tid & 31;
  const unsigned rowb = (tid >> 5) * 4;
  const unsigned ldsb =
      (unsigned)(unsigned long long)(&Bs[0][0][0][0]) +
      ((u >> 1) * TN + rowb) * 32 + (u & 1) * 16;
  const unsigned long long cbp = (unsigned long long)cb_bf_u;

  auto stage_async = [&](int tile, int buf) {
    unsigned gofs = (unsigned)tile * TILE_BYTES + rowb * 512 + u * 16;
    unsigned lofs = ldsb + (unsigned)buf * TILE_BYTES;
#pragma unroll
    for (int i = 0; i < 4; ++i) {
      asm volatile("global_load_async_to_lds_b128 %0, %1, %2"
                   :
                   : "v"(lofs + i * 32), "v"(gofs + i * 512), "s"(cbp)
                   : "memory");
    }
  };

  // ---- A tile: lane (half,rl) loads row (rbase+rl); per 32-K step it holds
  // K in {koff..koff+15}, koff = 16*half (contiguous, matches K-major B).
  const float* xr   = x + (size_t)(rbase + rl) * DIM;
  const int    koff = half * 16;
  v16bf A[8];
  float psum = 0.f;
#pragma unroll
  for (int ks = 0; ks < 8; ++ks) {
    const float* p0 = xr + ks * 32 + koff;
    float4 f0 = *(const float4*)(p0);
    float4 f1 = *(const float4*)(p0 + 4);
    float4 f2 = *(const float4*)(p0 + 8);
    float4 f3 = *(const float4*)(p0 + 12);
    psum += f0.x * f0.x + f0.y * f0.y + f0.z * f0.z + f0.w * f0.w;
    psum += f1.x * f1.x + f1.y * f1.y + f1.z * f1.z + f1.w * f1.w;
    psum += f2.x * f2.x + f2.y * f2.y + f2.z * f2.z + f2.w * f2.w;
    psum += f3.x * f3.x + f3.y * f3.y + f3.z * f3.z + f3.w * f3.w;
    v16bf a;
    a[0]  = (__bf16)f0.x; a[1]  = (__bf16)f0.y; a[2]  = (__bf16)f0.z; a[3]  = (__bf16)f0.w;
    a[4]  = (__bf16)f1.x; a[5]  = (__bf16)f1.y; a[6]  = (__bf16)f1.z; a[7]  = (__bf16)f1.w;
    a[8]  = (__bf16)f2.x; a[9]  = (__bf16)f2.y; a[10] = (__bf16)f2.z; a[11] = (__bf16)f2.w;
    a[12] = (__bf16)f3.x; a[13] = (__bf16)f3.y; a[14] = (__bf16)f3.z; a[15] = (__bf16)f3.w;
    A[ks] = a;
  }
  float fullsq = psum + __shfl_xor(psum, 16, 32);
  // C-layout: lanes 0-15 hold rows rbase+0..7; lanes 16-31 rows rbase+8..15.
  float xsqv[8];
#pragma unroll
  for (int j = 0; j < 8; ++j) xsqv[j] = __shfl(fullsq, half * 8 + j, 32);

  float minv[8];
  int   mini[8];
#pragma unroll
  for (int j = 0; j < 8; ++j) { minv[j] = 3.4e38f; mini[j] = 0; }

  // ---- prologue: async-fill tile 0 ----
  stage_async(0, 0);
  asm volatile("s_wait_asynccnt 0x0" ::: "memory");
  __syncthreads();

  for (int t = 0; t < NTILES; ++t) {
    const int buf = t & 1;
    if (t + 1 < NTILES) stage_async(t + 1, buf ^ 1);  // overlaps WMMA below

    const int nb = t * TN;
#pragma unroll
    for (int nt = 0; nt < TN / 16; ++nt) {
      // B fragment: lane reads 16 contiguous bf16 at [cc=ks*2+half][nt*16+rl]
      const __bf16* bbase = &Bs[buf][half][nt * 16 + rl][0];
      v8f acc = {0.f, 0.f, 0.f, 0.f, 0.f, 0.f, 0.f, 0.f};
#pragma unroll
      for (int ks = 0; ks < 8; ++ks) {
        v16bf b = *(const v16bf*)(bbase + ks * 1024);  // +2048 B per K-step
        acc = __builtin_amdgcn_wmma_f32_16x16x32_bf16(
            false, A[ks], false, b, (short)0, acc, false, false);
      }
      const int   cand = nb + nt * 16 + rl;
      const float cs   = csq_s[cand];
#pragma unroll
      for (int j = 0; j < 8; ++j) {
        float d2 = __builtin_fmaf(-2.f, acc[j], xsqv[j] + cs);
        if (d2 < minv[j]) { minv[j] = d2; mini[j] = cand; }
      }
    }

    asm volatile("s_wait_asynccnt 0x0" ::: "memory");  // next tile landed in LDS
    __syncthreads();
  }

  // min-reduce across the 16 lanes of each half (tie -> lowest index, like argmin)
#pragma unroll
  for (int j = 0; j < 8; ++j) {
#pragma unroll
    for (int s = 8; s > 0; s >>= 1) {
      float ov = __shfl_xor(minv[j], s, 32);
      int   oi = __shfl_xor(mini[j], s, 32);
      if (ov < minv[j] || (ov == minv[j] && oi < mini[j])) { minv[j] = ov; mini[j] = oi; }
    }
  }

  if (rl == 0) {
#pragma unroll
    for (int j = 0; j < 8; ++j)
      out_idx[rbase + half * 8 + j] = (float)mini[j];
  }

  // gather quantized rows from the ORIGINAL f32 codebook (exact output)
#pragma unroll
  for (int i = 0; i < 16; ++i) {
    int idx = __shfl(mini[i & 7], (i >> 3) * 16, 32);  // rows 0-7 lane 0, 8-15 lane 16
    const float4* src = (const float4*)(cb + (size_t)idx * DIM);
    float4*       dst = (float4*)(out_q + (size_t)(rbase + i) * DIM);
    dst[lane]      = src[lane];
    dst[lane + 32] = src[lane + 32];
  }
}

// ---------------------------------------------------------------------------
extern "C" void kernel_launch(void* const* d_in, const int* in_sizes, int n_in,
                              void* d_out, int out_size, void* d_ws, size_t ws_size,
                              hipStream_t stream) {
  (void)in_sizes; (void)n_in; (void)out_size; (void)ws_size;
  const float* x  = (const float*)d_in[0];
  const float* cb = (const float*)d_in[1];

  unsigned short* cb_bf = (unsigned short*)d_ws;                       // 4 MB
  float* csq = (float*)((char*)d_ws + (size_t)KCODES * DIM * sizeof(unsigned short));

  float* out_idx = (float*)d_out;          // [32768] indices (as float)
  float* out_q   = out_idx + MTOT;         // [32768 x 256] quantized f32

  vq_prep<<<KCODES / 8, 256, 0, stream>>>(cb, cb_bf, csq);
  vq_main<<<MTOT / TM, 256, 0, stream>>>(x, cb, cb_bf, csq, out_idx, out_q);
}
